// Encoder_31344671326348
// MI455X (gfx1250) — matrix operand order repaired and checked by reading, not compiled, not run on hardware
//
#include <hip/hip_runtime.h>
#include <hip/hip_bf16.h>

// Problem sizes (fixed by the reference)
#define BATCH 256
#define SEQ   2048
#define CHN   32
#define EMBD  2048

// Tiling: block 128x128, 8 waves as 4(m) x 2(n), wave tile 32x64
#define TM 128
#define TN 128
#define TK 32
#define LDR 40    // padded LDS row length in elems (80B rows; 16B-aligned chunks)

typedef __attribute__((ext_vector_type(16))) __bf16 bf16x16;
typedef __attribute__((ext_vector_type(8)))  __bf16 bf16x8;
typedef __attribute__((ext_vector_type(8)))  float  f32x8;

__device__ __forceinline__ bf16x16 cat16(bf16x8 lo, bf16x8 hi) {
    return __builtin_shufflevector(lo, hi, 0,1,2,3,4,5,6,7,8,9,10,11,12,13,14,15);
}

__launch_bounds__(256)
__global__ void grouped_linear_relu_bf16wmma(const float* __restrict__ x,
                                             const float* __restrict__ W,
                                             const float* __restrict__ bias,
                                             float* __restrict__ out) {
    __shared__ __bf16 As[TM][LDR];   // [m][k]
    __shared__ __bf16 Bs[TN][LDR];   // [n][k]  (transposed tile)

    const int tid = threadIdx.x;
    const int n0  = blockIdx.x * TN;
    const int m0  = blockIdx.y * TM;
    const int c   = blockIdx.z;

    const int lane = tid & 31;
    const int wave = tid >> 5;
    const int wm   = wave >> 1;     // 0..3  (M direction, 32 rows each)
    const int wn   = wave & 1;      // 0..1  (N direction, 64 cols each)
    const int lh   = lane >> 4;     // half-wave select
    const int ll   = lane & 15;

    // --- global load mappings ---
    // A tile: 128 rows (m) x 32 (k) fp32 from x, stride CHN along k
    const int ar = tid >> 1;            // 0..127
    const int ak = (tid & 1) * 16;      // 0 or 16
    // B tile: 32 rows (k) x 128 (e) fp32 from W, e coalesced across lanes
    const int be = tid & 127;           // 0..127
    const int bk = (tid >> 7) * 16;     // 0 or 16

    const float* xg = x + ((size_t)(m0 + ar) * SEQ + ak) * CHN + c;
    const float* wg = W + ((size_t)c * SEQ + bk) * EMBD + (n0 + be);

    f32x8 acc[2][4];
    #pragma unroll
    for (int i = 0; i < 2; ++i)
        #pragma unroll
        for (int j = 0; j < 4; ++j) acc[i][j] = (f32x8){};

    float aReg[16];
    float bReg[16];

    // initial global fetch (k = 0)
    #pragma unroll
    for (int j = 0; j < 16; ++j) aReg[j] = xg[(size_t)j * CHN];
    #pragma unroll
    for (int j = 0; j < 16; ++j) bReg[j] = wg[(size_t)j * EMBD];

    const int KT = SEQ / TK;  // 64
    for (int kt = 0; kt < KT; ++kt) {
        __syncthreads();   // previous compute done reading LDS
        // ---- stage current K-tile into LDS as bf16 (hardware cvt) ----
        {
            bf16x8 av0, av1, bv0, bv1;
            #pragma unroll
            for (int j = 0; j < 8; ++j) {
                av0[j] = (__bf16)aReg[j];
                av1[j] = (__bf16)aReg[8 + j];
                bv0[j] = (__bf16)bReg[j];
                bv1[j] = (__bf16)bReg[8 + j];
            }
            *(bf16x8*)&As[ar][ak]     = av0;
            *(bf16x8*)&As[ar][ak + 8] = av1;
            *(bf16x8*)&Bs[be][bk]     = bv0;   // transposed: [n][k]
            *(bf16x8*)&Bs[be][bk + 8] = bv1;
        }
        __syncthreads();

        // ---- prefetch next K-tile into registers (pipelined with WMMA) ----
        if (kt + 1 < KT) {
            const float* xn = xg + (size_t)(kt + 1) * TK * CHN;
            const float* wp = wg + (size_t)(kt + 1) * TK * EMBD;
            __builtin_prefetch(wp, 0, 1);   // global_prefetch_b8 for next W tile
            #pragma unroll
            for (int j = 0; j < 16; ++j) aReg[j] = xn[(size_t)j * CHN];
            #pragma unroll
            for (int j = 0; j < 16; ++j) bReg[j] = wp[(size_t)j * EMBD];
        }

        // ---- gather WMMA fragments from LDS (ISA 7.12.2 layouts) ----
        // A (16-bit 16x32): lanes 0-15 hold K {0-7,16-23}; lanes 16-31 K {8-15,24-31}
        bf16x16 afrag[2];
        #pragma unroll
        for (int ti = 0; ti < 2; ++ti) {
            const int am = wm * 32 + ti * 16 + ll;
            bf16x8 lo = *(const bf16x8*)&As[am][lh * 8];
            bf16x8 hi = *(const bf16x8*)&As[am][lh * 8 + 16];
            afrag[ti] = cat16(lo, hi);
        }
        // B (16-bit 32x16): lane holds column N=ll, K {0-7,16-23} (lh=0) / {8-15,24-31} (lh=1)
        bf16x16 bfrag[4];
        #pragma unroll
        for (int tj = 0; tj < 4; ++tj) {
            const int bn = wn * 64 + tj * 16 + ll;
            bf16x8 lo = *(const bf16x8*)&Bs[bn][lh * 8];
            bf16x8 hi = *(const bf16x8*)&Bs[bn][lh * 8 + 16];
            bfrag[tj] = cat16(lo, hi);
        }

        #pragma unroll
        for (int ti = 0; ti < 2; ++ti)
            #pragma unroll
            for (int tj = 0; tj < 4; ++tj)
                acc[ti][tj] = __builtin_amdgcn_wmma_f32_16x16x32_bf16(
                    false, afrag[ti], false, bfrag[tj], (short)0, acc[ti][tj],
                    false, false);
    }

    // ---- epilogue: bias + ReLU, store ----
    // C/D layout: VGPR i -> M = i (lanes 0-15) or M = i+8 (lanes 16-31); N = ll
    #pragma unroll
    for (int tj = 0; tj < 4; ++tj) {
        const int ng = n0 + wn * 64 + tj * 16 + ll;
        const float bq = bias[(size_t)c * EMBD + ng];
        #pragma unroll
        for (int ti = 0; ti < 2; ++ti) {
            const int mb = m0 + wm * 32 + ti * 16 + lh * 8;
            #pragma unroll
            for (int i = 0; i < 8; ++i) {
                const int r = mb + i;
                out[((size_t)r * CHN + c) * EMBD + ng] =
                    fmaxf(acc[ti][tj][i] + bq, 0.0f);
            }
        }
    }
}

extern "C" void kernel_launch(void* const* d_in, const int* in_sizes, int n_in,
                              void* d_out, int out_size, void* d_ws, size_t ws_size,
                              hipStream_t stream) {
    (void)in_sizes; (void)n_in; (void)out_size; (void)d_ws; (void)ws_size;
    const float* x  = (const float*)d_in[0];   // (B, S, C)
    const float* W  = (const float*)d_in[1];   // (C, S, E)
    const float* b  = (const float*)d_in[2];   // (C, E)
    float*       o  = (float*)d_out;           // (B, C, E)

    dim3 grid(EMBD / TN, BATCH / TM, CHN);     // (16, 2, 32)
    grouped_linear_relu_bf16wmma<<<grid, 256, 0, stream>>>(x, W, b, o);
}
